// TurbulentExpert_72060961292572
// MI455X (gfx1250) — matrix operand order repaired and checked by reading, not compiled, not run on hardware
//
#include <hip/hip_runtime.h>
#include <hip/hip_bf16.h>
#include <math.h>

// ---------------------------------------------------------------------------
// GAT (4 layers, heads=4, HID=256) for MI455X / gfx1250.
// Big GEMMs via v_wmma_f32_16x16x32_f16 (f16 in, f32 accum).
// Edge phase: segment-softmax via ordered-uint atomicMax + f32 atomicAdd.
// ---------------------------------------------------------------------------

typedef __attribute__((ext_vector_type(16))) _Float16     v16h;
typedef __attribute__((ext_vector_type(8)))  float        v8f;
typedef __attribute__((ext_vector_type(4)))  unsigned int v4u;

#define HIDC  256
#define HEADS 4
#define HEADC 64
#define INCH  16
#define OUTCH 4

union Frag16 { v16h h; v4u u[2]; };

// ---------------------------------------------------------------------------
// Weight convert + transpose: whT[l][col][k] = f16(w_gat[l][k][col])
// ---------------------------------------------------------------------------
__global__ void k_wcvt(const float* __restrict__ w, _Float16* __restrict__ wt) {
  int t = blockIdx.x * blockDim.x + threadIdx.x;
  if (t >= 4 * HIDC * HIDC) return;
  int l   = t >> 16;
  int rem = t & 0xFFFF;
  int col = rem >> 8;
  int k   = rem & 255;
  wt[t] = (_Float16)w[(size_t)l * HIDC * HIDC + (size_t)k * HIDC + col];
}

// ---------------------------------------------------------------------------
// Input projection: x0 = x @ w_in + b_in  ([N,16]@[16,256]); block = node
// ---------------------------------------------------------------------------
__global__ __launch_bounds__(256) void k_inproj(
    const float* __restrict__ x, const float* __restrict__ w,
    const float* __restrict__ b, float* __restrict__ x0,
    _Float16* __restrict__ hf16, int nNodes) {
  __shared__ float xs[INCH];
  int nd = blockIdx.x;
  if (nd >= nNodes) return;
  int c = threadIdx.x;
  if (c < INCH) xs[c] = x[(size_t)nd * INCH + c];
  __syncthreads();
  float s = b[c];
#pragma unroll
  for (int k = 0; k < INCH; ++k) s += xs[k] * w[k * HIDC + c];
  size_t o = (size_t)nd * HIDC + c;
  x0[o]   = s;
  hf16[o] = (_Float16)s;
}

// ---------------------------------------------------------------------------
// C[N,256] = A_f16[N,256] @ W (via BT = W^T f16).  Wave-per 16x64 C tile,
// 4 x v_wmma_f32_16x16x32_f16 per k-step, 8 k-steps.
// A frag (16-bit 16x32): lane m=lane&15, kbase = lane<16 ? 0 : 8;
//   halves 0..7 = K kbase..kbase+7, halves 8..15 = K 16+kbase..16+kbase+7.
// B frag (32x16): lane n=lane&15, koff = lane<16 ? 0 : 8;
//   halves 0..7 = K koff..koff+7, halves 8..15 = K 16+koff..16+koff+7 of column n.
// ---------------------------------------------------------------------------
__global__ __launch_bounds__(256) void k_gemm256(
    const _Float16* __restrict__ A, const _Float16* __restrict__ BT,
    float* __restrict__ C, int nRows) {
  const int wave = (int)((blockIdx.x * blockDim.x + threadIdx.x) >> 5);
  const int lane = threadIdx.x & 31;
  const int nRowTiles = (nRows + 15) >> 4;
  const int rowTile  = wave >> 2;
  const int colGroup = wave & 3;
  if (rowTile >= nRowTiles) return;
  const int rowBase = rowTile << 4;
  const int m     = lane & 15;
  const int kbase = (lane < 16) ? 0 : 8;
  const int koff  = (lane < 16) ? 0 : 8;
  int arow = rowBase + m;
  if (arow >= nRows) arow = nRows - 1;
  const _Float16* Abase = A + (size_t)arow * HIDC + kbase;

  v8f acc[4] = {};
#pragma unroll
  for (int ks = 0; ks < 8; ++ks) {
    const int k0 = ks * 32;
    if (ks < 7) __builtin_prefetch(Abase + k0 + 32, 0, 1);
    Frag16 fa;
    fa.u[0] = *(const v4u*)(Abase + k0);
    fa.u[1] = *(const v4u*)(Abase + k0 + 16);
#pragma unroll
    for (int t = 0; t < 4; ++t) {
      const int col0 = (colGroup << 6) + (t << 4);
      const _Float16* Bp = BT + (size_t)(col0 + m) * HIDC + k0 + koff;
      Frag16 fb;
      fb.u[0] = *(const v4u*)(Bp);
      fb.u[1] = *(const v4u*)(Bp + 16);
      acc[t] = __builtin_amdgcn_wmma_f32_16x16x32_f16(
          false, fa.h, false, fb.h, (short)0, acc[t], false, false);
    }
  }
  // C/D layout: VGPR r -> row (r + (lane<16?0:8)), col = lane&15 within tile
  const int rOff = (lane < 16) ? 0 : 8;
#pragma unroll
  for (int t = 0; t < 4; ++t) {
    const int col = (colGroup << 6) + (t << 4) + m;
#pragma unroll
    for (int r = 0; r < 8; ++r) {
      const int row = rowBase + r + rOff;
      if (row < nRows) C[(size_t)row * HIDC + col] = acc[t][r];
    }
  }
}

// ---------------------------------------------------------------------------
// Attention scalars: a_s[n,h], a_d[n,h] = <hw[n,h,:], att_{src,dst}[h,:]>
// ---------------------------------------------------------------------------
__global__ void k_att(const float* __restrict__ hw,
                      const float* __restrict__ asrc,
                      const float* __restrict__ adst,
                      float* __restrict__ outs, float* __restrict__ outd,
                      int nNodes) {
  int t = blockIdx.x * blockDim.x + threadIdx.x;
  if (t >= nNodes * HEADS) return;
  int nd = t >> 2, hd = t & 3;
  const float* hp = hw + (size_t)nd * HIDC + hd * HEADC;
  const float* s  = asrc + hd * HEADC;
  const float* d  = adst + hd * HEADC;
  float ss = 0.f, sd = 0.f;
#pragma unroll
  for (int c = 0; c < HEADC; c += 4) {
    float4 hv = *(const float4*)(hp + c);
    float4 sv = *(const float4*)(s + c);
    float4 dv = *(const float4*)(d + c);
    ss += hv.x * sv.x + hv.y * sv.y + hv.z * sv.z + hv.w * sv.w;
    sd += hv.x * dv.x + hv.y * dv.y + hv.z * dv.z + hv.w * dv.w;
  }
  outs[t] = ss;
  outd[t] = sd;
}

// ordered-uint encoding: monotonic float -> uint map (max works for negatives)
__device__ __forceinline__ unsigned enc_f(float f) {
  unsigned b = __float_as_uint(f);
  return (b & 0x80000000u) ? ~b : (b | 0x80000000u);
}
__device__ __forceinline__ float dec_f(unsigned u) {
  unsigned b = (u & 0x80000000u) ? (u & 0x7FFFFFFFu) : ~u;
  return __uint_as_float(b);
}

__device__ __forceinline__ void edge_sd(const int* ei, int E, int e, int& s, int& d) {
  if (e < E) { s = ei[e]; d = ei[E + e]; }
  else       { s = d = e - E; }          // appended self-loops
}

// ---------------------------------------------------------------------------
// Edge pass 1: logit = leakyrelu(a_s[src]+a_d[dst]); segment max into mmax
// ---------------------------------------------------------------------------
__global__ void k_edge_logit_max(const int* __restrict__ ei, int E, int Etot,
                                 const float* __restrict__ as,
                                 const float* __restrict__ ad,
                                 float* __restrict__ logits,
                                 unsigned* __restrict__ mmax) {
  int t = blockIdx.x * blockDim.x + threadIdx.x;
  if (t >= Etot * HEADS) return;
  int e = t >> 2, hd = t & 3;
  int s, d; edge_sd(ei, E, e, s, d);
  float l = as[(size_t)s * HEADS + hd] + ad[(size_t)d * HEADS + hd];
  l = (l > 0.f) ? l : 0.2f * l;
  logits[t] = l;
  atomicMax(mmax + (size_t)d * HEADS + hd, enc_f(l));
}

// ---------------------------------------------------------------------------
// Edge pass 2: e = exp(logit - m[dst]); segment sum into den
// ---------------------------------------------------------------------------
__global__ void k_edge_exp_sum(const int* __restrict__ ei, int E, int Etot,
                               float* __restrict__ evals,
                               const unsigned* __restrict__ mmax,
                               float* __restrict__ den) {
  int t = blockIdx.x * blockDim.x + threadIdx.x;
  if (t >= Etot * HEADS) return;
  int e = t >> 2, hd = t & 3;
  int s, d; edge_sd(ei, E, e, s, d);
  float ev = expf(evals[t] - dec_f(mmax[(size_t)d * HEADS + hd]));
  evals[t] = ev;
  atomicAdd(den + (size_t)d * HEADS + hd, ev);
}

// ---------------------------------------------------------------------------
// Edge pass 3: agg[dst] += alpha * hw[src].  Wave per edge, 8 ch / lane.
// ---------------------------------------------------------------------------
__global__ __launch_bounds__(256) void k_edge_msg(
    const int* __restrict__ ei, int E, int Etot,
    const float* __restrict__ evals, const float* __restrict__ den,
    const float* __restrict__ hw, float* __restrict__ agg) {
  int wave = (int)((blockIdx.x * blockDim.x + threadIdx.x) >> 5);
  int lane = threadIdx.x & 31;
  if (wave >= Etot) return;
  int s, d; edge_sd(ei, E, wave, s, d);
  float alpha[HEADS];
#pragma unroll
  for (int h = 0; h < HEADS; ++h)
    alpha[h] = evals[(size_t)wave * HEADS + h] /
               (den[(size_t)d * HEADS + h] + 1e-16f);
  const float* hs = hw + (size_t)s * HIDC;
  float* ag = agg + (size_t)d * HIDC;
#pragma unroll
  for (int i = 0; i < 8; ++i) {
    int c = lane + 32 * i;
    atomicAdd(ag + c, alpha[c >> 6] * hs[c]);
  }
}

// ---------------------------------------------------------------------------
// Bias + LayerNorm + ELU; writes f32 next-h and f16 copy. Wave per node.
// ---------------------------------------------------------------------------
__global__ __launch_bounds__(256) void k_ln_elu(
    const float* __restrict__ agg, const float* __restrict__ bias,
    const float* __restrict__ g, const float* __restrict__ beta,
    float* __restrict__ hout, _Float16* __restrict__ hf16, int nNodes) {
  int wave = (int)((blockIdx.x * blockDim.x + threadIdx.x) >> 5);
  int lane = threadIdx.x & 31;
  if (wave >= nNodes) return;
  const float* a = agg + (size_t)wave * HIDC;
  float v[8], s = 0.f, sq = 0.f;
#pragma unroll
  for (int i = 0; i < 8; ++i) {
    int c = lane + 32 * i;
    v[i] = a[c] + bias[c];
    s += v[i];
    sq += v[i] * v[i];
  }
#pragma unroll
  for (int o = 16; o >= 1; o >>= 1) {
    s  += __shfl_xor(s, o, 32);
    sq += __shfl_xor(sq, o, 32);
  }
  float mean = s * (1.f / HIDC);
  float var  = sq * (1.f / HIDC) - mean * mean;
  float r    = rsqrtf(var + 1e-5f);
#pragma unroll
  for (int i = 0; i < 8; ++i) {
    int c = lane + 32 * i;
    float y = (v[i] - mean) * r * g[c] + beta[c];
    y = (y > 0.f) ? y : expm1f(y);
    size_t o = (size_t)wave * HIDC + c;
    hout[o] = y;
    hf16[o] = (_Float16)y;
  }
}

// ---------------------------------------------------------------------------
// Residual add: x += h; refresh f16 copy
// ---------------------------------------------------------------------------
__global__ void k_resid(float* __restrict__ x, const float* __restrict__ h,
                        _Float16* __restrict__ hf16, size_t n) {
  size_t t = (size_t)blockIdx.x * blockDim.x + threadIdx.x;
  if (t >= n) return;
  float v = x[t] + h[t];
  x[t]    = v;
  hf16[t] = (_Float16)v;
}

// ---------------------------------------------------------------------------
// Output projection: [N,256]@[256,4] + b_out. Thread per node.
// ---------------------------------------------------------------------------
__global__ void k_outproj(const float* __restrict__ x,
                          const float* __restrict__ w,
                          const float* __restrict__ b,
                          float* __restrict__ out, int nNodes) {
  int nd = blockIdx.x * blockDim.x + threadIdx.x;
  if (nd >= nNodes) return;
  const float* xp = x + (size_t)nd * HIDC;
  float4 acc = *(const float4*)b;
  for (int k = 0; k < HIDC; k += 4) {
    float4 xv = *(const float4*)(xp + k);
    float4 w0 = *(const float4*)(w + (k + 0) * OUTCH);
    float4 w1 = *(const float4*)(w + (k + 1) * OUTCH);
    float4 w2 = *(const float4*)(w + (k + 2) * OUTCH);
    float4 w3 = *(const float4*)(w + (k + 3) * OUTCH);
    acc.x += xv.x * w0.x + xv.y * w1.x + xv.z * w2.x + xv.w * w3.x;
    acc.y += xv.x * w0.y + xv.y * w1.y + xv.z * w2.y + xv.w * w3.y;
    acc.z += xv.x * w0.z + xv.y * w1.z + xv.z * w2.z + xv.w * w3.z;
    acc.w += xv.x * w0.w + xv.y * w1.w + xv.z * w2.w + xv.w * w3.w;
  }
  *(float4*)(out + (size_t)nd * OUTCH) = acc;
}

// ---------------------------------------------------------------------------
extern "C" void kernel_launch(void* const* d_in, const int* in_sizes, int n_in,
                              void* d_out, int out_size, void* d_ws, size_t ws_size,
                              hipStream_t stream) {
  const float* x_in     = (const float*)d_in[0];
  const int*   ei       = (const int*)d_in[1];
  const float* w_in     = (const float*)d_in[2];
  const float* b_in     = (const float*)d_in[3];
  const float* w_gat    = (const float*)d_in[4];
  const float* att_src  = (const float*)d_in[5];
  const float* att_dst  = (const float*)d_in[6];
  const float* b_gat    = (const float*)d_in[7];
  const float* ln_scale = (const float*)d_in[8];
  const float* ln_bias  = (const float*)d_in[9];
  const float* w_out    = (const float*)d_in[10];
  const float* b_out    = (const float*)d_in[11];

  const int N    = in_sizes[0] / INCH;
  const int E    = in_sizes[1] / 2;
  const int Etot = E + N;                     // + self-loops

  // ---- workspace carve-out ----
  char* p = (char*)d_ws;
  auto carve = [&](size_t bytes) -> void* {
    void* r = (void*)p;
    p += (bytes + 255) & ~(size_t)255;
    return r;
  };
  const size_t szF = (size_t)N * HIDC * sizeof(float);
  float*     buf_x    = (float*)carve(szF);                     // residual base
  float*     buf_h    = (float*)carve(szF);                     // layer output (f32)
  float*     buf_hw   = (float*)carve(szF);                     // h = x @ W
  float*     buf_agg  = (float*)carve(szF);                     // message aggregation
  _Float16*  buf_hf16 = (_Float16*)carve((size_t)N * HIDC * 2); // GEMM A operand
  _Float16*  whT      = (_Float16*)carve((size_t)4 * HIDC * HIDC * 2);
  float*     buf_as   = (float*)carve((size_t)N * HEADS * 4);
  float*     buf_ad   = (float*)carve((size_t)N * HEADS * 4);
  unsigned*  buf_m    = (unsigned*)carve((size_t)N * HEADS * 4);
  float*     buf_den  = (float*)carve((size_t)N * HEADS * 4);
  float*     buf_e    = (float*)carve((size_t)Etot * HEADS * 4);

  // ---- one-time weight convert/transpose ----
  {
    int tot = 4 * HIDC * HIDC;
    k_wcvt<<<(tot + 255) / 256, 256, 0, stream>>>(w_gat, whT);
  }

  // ---- input projection ----
  k_inproj<<<N, HIDC, 0, stream>>>(x_in, w_in, b_in, buf_x, buf_hf16, N);

  const int nRowTiles  = (N + 15) / 16;
  const int gemmWaves  = nRowTiles * 4;
  const int gemmBlocks = (gemmWaves * 32 + 255) / 256;
  const int attBlocks  = (N * HEADS + 255) / 256;
  const int ehBlocks   = (Etot * HEADS + 255) / 256;
  const int msgBlocks  = (Etot + 7) / 8;      // wave per edge, 8 waves/block
  const int lnBlocks   = (N + 7) / 8;         // wave per node

  auto gat_layer = [&](int l) {
    k_gemm256<<<gemmBlocks, 256, 0, stream>>>(
        buf_hf16, whT + (size_t)l * HIDC * HIDC, buf_hw, N);
    k_att<<<attBlocks, 256, 0, stream>>>(
        buf_hw, att_src + (size_t)l * HIDC, att_dst + (size_t)l * HIDC,
        buf_as, buf_ad, N);
    hipMemsetAsync(buf_m,   0, (size_t)N * HEADS * 4, stream); // enc(-NaN) floor
    hipMemsetAsync(buf_den, 0, (size_t)N * HEADS * 4, stream);
    hipMemsetAsync(buf_agg, 0, szF, stream);
    k_edge_logit_max<<<ehBlocks, 256, 0, stream>>>(
        ei, E, Etot, buf_as, buf_ad, buf_e, buf_m);
    k_edge_exp_sum<<<ehBlocks, 256, 0, stream>>>(
        ei, E, Etot, buf_e, buf_m, buf_den);
    k_edge_msg<<<msgBlocks, 256, 0, stream>>>(
        ei, E, Etot, buf_e, buf_den, buf_hw, buf_agg);
    k_ln_elu<<<lnBlocks, 256, 0, stream>>>(
        buf_agg, b_gat + (size_t)l * HIDC, ln_scale + (size_t)l * HIDC,
        ln_bias + (size_t)l * HIDC, buf_h, buf_hf16, N);
  };

  const size_t nElem = (size_t)N * HIDC;
  const int residBlocks = (int)((nElem + 255) / 256);

  gat_layer(0);
  gat_layer(1);
  k_resid<<<residBlocks, 256, 0, stream>>>(buf_x, buf_h, buf_hf16, nElem);
  gat_layer(2);
  gat_layer(3);
  k_resid<<<residBlocks, 256, 0, stream>>>(buf_x, buf_h, buf_hf16, nElem);

  k_outproj<<<(N + 255) / 256, 256, 0, stream>>>(
      buf_x, w_out, b_out, (float*)d_out, N);
}
